// TransformerLayer_28630251995905
// MI455X (gfx1250) — compile-verified, matrix operand order checked by reading, not run on hardware
//
#include <hip/hip_runtime.h>
#include <hip/hip_bf16.h>
#include <math.h>

typedef __bf16 bf16_t;
typedef bf16_t v16bf __attribute__((ext_vector_type(16)));
typedef float  v8f   __attribute__((ext_vector_type(8)));

constexpr int Bn = 4;      // batch
constexpr int Sn = 2048;   // sequence
constexpr int Dn = 1024;   // hidden
constexpr int TILE_M = 128;
constexpr int TILE_N = 128;
constexpr int TILE_K = 64;                  // two WMMA k-slices per LDS stage
constexpr int ASTRIDE = 72;                 // halves per LDS row (64 + 8 pad); 144B rows, 16B aligned
constexpr int SMEM_HALVES = TILE_M * ASTRIDE;

// ---------------- CDNA5 async staging helpers -----------------

// One 128x64-half (16KB) tile: 1024 chunks of 8 halves; 256 threads x 4 chunks.
// Data goes straight to LDS via the async engine (ASYNCcnt), no VGPR staging.
__device__ __forceinline__ void async_tile(const bf16_t* __restrict__ g, int ld,
                                           int row0, int k0, uint32_t lds_base) {
  int t = threadIdx.x;
#pragma unroll
  for (int i = 0; i < 4; ++i) {
    int c    = t + i * 256;        // 0..1023
    int row  = c >> 3;
    int part = c & 7;
    const void* gp = g + (size_t)(row0 + row) * ld + k0 + part * 8;
    uint32_t lo = lds_base + (uint32_t)(row * ASTRIDE + part * 8) * 2u;
    asm volatile("global_load_async_to_lds_b128 %0, %1, off"
                 :: "v"(lo), "v"(gp) : "memory");
  }
}

__device__ __forceinline__ void wait_async0() {
  asm volatile("s_wait_asynccnt 0x0" ::: "memory");
}

__device__ __forceinline__ v16bf load_frag(const bf16_t* __restrict__ sm, int row0, int kofs) {
  int lane = threadIdx.x & 31;
  int r  = row0 + (lane & 15);
  int kb = kofs + (lane >> 4) * 8;   // half-element offset within the 64-wide stage
  union { v16bf v; uint4 u[2]; } f;
  f.u[0] = *(const uint4*)(sm + r * ASTRIDE + kb);
  f.u[1] = *(const uint4*)(sm + r * ASTRIDE + 16 + kb);
  return f.v;
}

// ---------------- GEMM core (bf16 x bf16 -> f32, WMMA 16x16x32) -----------------
// 256-thread block = 8 waves computes a 128x128 tile.  Wave (wm,wn) owns 32x64:
// 2 M-subtiles x 4 N-subtiles.  Double-buffered async LDS staging, K-step 64.
__device__ __forceinline__ void gemm_core(const bf16_t* __restrict__ A,
                                          const bf16_t* __restrict__ Bt,
                                          int K, int m0, int n0,
                                          v8f acc[2][4]) {
  __shared__ bf16_t sA[2][SMEM_HALVES];
  __shared__ bf16_t sB[2][SMEM_HALVES];
  uint32_t aBase[2] = {(uint32_t)(uintptr_t)&sA[0][0], (uint32_t)(uintptr_t)&sA[1][0]};
  uint32_t bBase[2] = {(uint32_t)(uintptr_t)&sB[0][0], (uint32_t)(uintptr_t)&sB[1][0]};

  int wave = threadIdx.x >> 5;
  int wm = wave >> 1;   // 0..3
  int wn = wave & 1;    // 0..1
  v8f zero = {0.f, 0.f, 0.f, 0.f, 0.f, 0.f, 0.f, 0.f};
#pragma unroll
  for (int i = 0; i < 2; ++i)
#pragma unroll
    for (int j = 0; j < 4; ++j) acc[i][j] = zero;

  // Prologue: stage tile 0 into buffer 0.
  async_tile(A,  K, m0, 0, aBase[0]);
  async_tile(Bt, K, n0, 0, bBase[0]);
  wait_async0();
  __syncthreads();

  int nk = K / TILE_K;
  for (int kk = 0; kk < nk; ++kk) {
    int cur = kk & 1, nxt = cur ^ 1;
    // Prefetch next stage into the other buffer (its readers finished last iter).
    if (kk + 1 < nk) {
      async_tile(A,  K, m0, (kk + 1) * TILE_K, aBase[nxt]);
      async_tile(Bt, K, n0, (kk + 1) * TILE_K, bBase[nxt]);
    }
    const bf16_t* cA = sA[cur];
    const bf16_t* cB = sB[cur];
#pragma unroll
    for (int kf = 0; kf < 2; ++kf) {          // two 16x16x32 k-slices
      v16bf af[2], bfr[4];
#pragma unroll
      for (int i = 0; i < 2; ++i) af[i]  = load_frag(cA, wm * 32 + i * 16, kf * 32);
#pragma unroll
      for (int j = 0; j < 4; ++j) bfr[j] = load_frag(cB, wn * 64 + j * 16, kf * 32);
#pragma unroll
      for (int i = 0; i < 2; ++i)
#pragma unroll
        for (int j = 0; j < 4; ++j)
          acc[i][j] = __builtin_amdgcn_wmma_f32_16x16x32_bf16(
              false, af[i], false, bfr[j], (short)0, acc[i][j], false, false);
    }
    // Next buffer must be resident (all waves) before anyone reads it.
    wait_async0();
    __syncthreads();
  }
}

// ---------------- conversion / prep kernels -----------------

__global__ void cvt_src_kernel(const float* __restrict__ src, bf16_t* __restrict__ dst, int n) {
  int i = blockIdx.x * blockDim.x + threadIdx.x;
  if (i < n) dst[i] = (bf16_t)src[i];
}

// Wt[w][n][k] = W_w[k][n]   (bf16, transposed weights, 3 matrices concatenated)
__global__ void cvt_wt_kernel(const float* __restrict__ Wq, const float* __restrict__ Wk,
                              const float* __restrict__ Wv, bf16_t* __restrict__ Wt) {
  __shared__ float tile[32][33];
  int w = blockIdx.z;
  const float* W = (w == 0) ? Wq : (w == 1) ? Wk : Wv;
  bf16_t* dst = Wt + (size_t)w * Dn * Dn;
  int k0 = blockIdx.x * 32, n0 = blockIdx.y * 32;
  int tx = threadIdx.x & 31, ty = threadIdx.x >> 5;   // 32 x 8
#pragma unroll
  for (int r = ty; r < 32; r += 8) tile[r][tx] = W[(size_t)(k0 + r) * Dn + n0 + tx];
  __syncthreads();
#pragma unroll
  for (int r = ty; r < 32; r += 8)
    dst[(size_t)(n0 + r) * Dn + k0 + tx] = (bf16_t)tile[tx][r];
}

// ---------------- QKV GEMM with "buggy transpose" permutation epilogue -----------------

__global__ void qkv_gemm_kernel(const bf16_t* __restrict__ srcb, const bf16_t* __restrict__ Wt,
                                bf16_t* __restrict__ Qp, bf16_t* __restrict__ Kp,
                                bf16_t* __restrict__ Vt) {
  int m0 = blockIdx.x * TILE_M;            // over B*S = 8192
  int n0 = blockIdx.y * TILE_N;            // over D
  int w  = blockIdx.z;                     // 0:Q 1:K 2:V
  const bf16_t* Bt = Wt + (size_t)w * Dn * Dn;
  v8f acc[2][4];
  gemm_core(srcb, Bt, Dn, m0, n0, acc);

  int lane = threadIdx.x & 31;
  int wave = threadIdx.x >> 5;
  int wm = wave >> 1, wn = wave & 1;
  bf16_t* dst = (w == 0) ? Qp : Kp;
#pragma unroll
  for (int i = 0; i < 2; ++i)
#pragma unroll
    for (int j = 0; j < 4; ++j)
#pragma unroll
      for (int v = 0; v < 8; ++v) {
        int m = m0 + wm * 32 + i * 16 + ((lane >> 4) << 3) + v;
        int n = n0 + wn * 64 + j * 16 + (lane & 15);
        int b = m >> 11;          // / 2048
        int s = m & 2047;
        int h = n >> 6;           // / 64
        int dl = n & 63;
        int sp = h * 128 + (s >> 4);
        int dp = ((s & 15) << 6) + dl;
        bf16_t val = (bf16_t)acc[i][j][v];
        if (w == 2)
          Vt[((size_t)b * Dn + dp) * Sn + sp] = val;        // V stored transposed (D x S)
        else
          dst[((size_t)b * Sn + sp) * Dn + dp] = val;       // Q/K permuted row-major
      }
}

// ---------------- scores = Qp . Kp^T / 32  (raw, pre-softmax, into `a` output region) ---

__global__ void scores_gemm_kernel(const bf16_t* __restrict__ Qp, const bf16_t* __restrict__ Kp,
                                   float* __restrict__ a_out) {
  int m0 = blockIdx.x * TILE_M;
  int n0 = blockIdx.y * TILE_N;
  int b  = blockIdx.z;
  const bf16_t* A  = Qp + (size_t)b * Sn * Dn;
  const bf16_t* Bt = Kp + (size_t)b * Sn * Dn;
  v8f acc[2][4];
  gemm_core(A, Bt, Dn, m0, n0, acc);

  int lane = threadIdx.x & 31;
  int wave = threadIdx.x >> 5;
  int wm = wave >> 1, wn = wave & 1;
  float* ab = a_out + (size_t)b * Sn * Sn;
#pragma unroll
  for (int i = 0; i < 2; ++i)
#pragma unroll
    for (int j = 0; j < 4; ++j)
#pragma unroll
      for (int v = 0; v < 8; ++v) {
        int m = m0 + wm * 32 + i * 16 + ((lane >> 4) << 3) + v;
        int n = n0 + wn * 64 + j * 16 + (lane & 15);
        ab[(size_t)m * Sn + n] = acc[i][j][v] * 0.03125f;   // 1/sqrt(1024)
      }
}

// ---------------- softmax over rows of `a` (in place) + bf16 probs -----------------

__global__ void softmax_kernel(float* __restrict__ a, bf16_t* __restrict__ pb) {
  __shared__ float red[256];
  size_t row = blockIdx.x;                 // B*S rows
  float*  p = a  + row * (size_t)Sn;
  bf16_t* q = pb + row * (size_t)Sn;
  int t = threadIdx.x;
  float vals[8];
  float mx = -INFINITY;
#pragma unroll
  for (int i = 0; i < 8; ++i) { vals[i] = p[t + i * 256]; mx = fmaxf(mx, vals[i]); }
  red[t] = mx; __syncthreads();
  for (int off = 128; off > 0; off >>= 1) {
    if (t < off) red[t] = fmaxf(red[t], red[t + off]);
    __syncthreads();
  }
  mx = red[0]; __syncthreads();
  float sum = 0.f;
#pragma unroll
  for (int i = 0; i < 8; ++i) { vals[i] = __expf(vals[i] - mx); sum += vals[i]; }
  red[t] = sum; __syncthreads();
  for (int off = 128; off > 0; off >>= 1) {
    if (t < off) red[t] += red[t + off];
    __syncthreads();
  }
  float inv = 1.0f / red[0];
#pragma unroll
  for (int i = 0; i < 8; ++i) {
    float r = vals[i] * inv;
    p[t + i * 256] = r;
    q[t + i * 256] = (bf16_t)r;
  }
}

// ---------------- att = probs . Vp   (B,S,D) fp32 into workspace -----------------

__global__ void att_gemm_kernel(const bf16_t* __restrict__ Pb, const bf16_t* __restrict__ Vt,
                                float* __restrict__ att) {
  int m0 = blockIdx.x * TILE_M;            // over S
  int n0 = blockIdx.y * TILE_N;            // over D
  int b  = blockIdx.z;
  const bf16_t* A  = Pb + (size_t)b * Sn * Sn;   // M=S, K=S
  const bf16_t* Bt = Vt + (size_t)b * Dn * Sn;   // N=D rows, K=S
  v8f acc[2][4];
  gemm_core(A, Bt, Sn, m0, n0, acc);

  int lane = threadIdx.x & 31;
  int wave = threadIdx.x >> 5;
  int wm = wave >> 1, wn = wave & 1;
  float* ob = att + (size_t)b * Sn * Dn;
#pragma unroll
  for (int i = 0; i < 2; ++i)
#pragma unroll
    for (int j = 0; j < 4; ++j)
#pragma unroll
      for (int v = 0; v < 8; ++v) {
        int m = m0 + wm * 32 + i * 16 + ((lane >> 4) << 3) + v;
        int n = n0 + wn * 64 + j * 16 + (lane & 15);
        ob[(size_t)m * Dn + n] = acc[i][j][v];
      }
}

// ---------------- fused LN1 + residual + LN2 -----------------

__device__ __forceinline__ float block_sum(float x, float* red) {
  int t = threadIdx.x;
  red[t] = x; __syncthreads();
  for (int off = 128; off > 0; off >>= 1) {
    if (t < off) red[t] += red[t + off];
    __syncthreads();
  }
  float r = red[0]; __syncthreads();
  return r;
}

__global__ void addnorm_kernel(const float* __restrict__ att, const float* __restrict__ src,
                               const float* __restrict__ ln1w, const float* __restrict__ ln1b,
                               const float* __restrict__ ln2w, const float* __restrict__ ln2b,
                               float* __restrict__ X) {
  __shared__ float red[256];
  size_t row = blockIdx.x;                 // B*S rows of length D
  const float* ar = att + row * (size_t)Dn;
  const float* sr = src + row * (size_t)Dn;
  float* xr = X + row * (size_t)Dn;
  int t = threadIdx.x;
  const float invD = 1.0f / (float)Dn;
  const float eps = 1e-5f;

  float v[4], s = 0.f;
#pragma unroll
  for (int i = 0; i < 4; ++i) { v[i] = ar[t + i * 256]; s += v[i]; }
  float mu = block_sum(s, red) * invD;
  float q = 0.f;
#pragma unroll
  for (int i = 0; i < 4; ++i) { float d = v[i] - mu; q += d * d; }
  float rstd = rsqrtf(block_sum(q, red) * invD + eps);

  float y[4]; s = 0.f;
#pragma unroll
  for (int i = 0; i < 4; ++i) {
    int c = t + i * 256;
    float src2 = (v[i] - mu) * rstd * ln1w[c] + ln1b[c];
    y[i] = sr[c] + src2;
    s += y[i];
  }
  float mu2 = block_sum(s, red) * invD;
  q = 0.f;
#pragma unroll
  for (int i = 0; i < 4; ++i) { float d = y[i] - mu2; q += d * d; }
  float rstd2 = rsqrtf(block_sum(q, red) * invD + eps);
#pragma unroll
  for (int i = 0; i < 4; ++i) {
    int c = t + i * 256;
    xr[c] = (y[i] - mu2) * rstd2 * ln2w[c] + ln2b[c];
  }
}

// ---------------- launcher -----------------

extern "C" void kernel_launch(void* const* d_in, const int* in_sizes, int n_in,
                              void* d_out, int out_size, void* d_ws, size_t ws_size,
                              hipStream_t stream) {
  const float* src  = (const float*)d_in[0];
  const float* Wq   = (const float*)d_in[1];
  const float* Wk   = (const float*)d_in[2];
  const float* Wv   = (const float*)d_in[3];
  const float* ln1w = (const float*)d_in[4];
  const float* ln1b = (const float*)d_in[5];
  const float* ln2w = (const float*)d_in[6];
  const float* ln2b = (const float*)d_in[7];

  float* X = (float*)d_out;                         // (B,S,D)
  float* a = X + (size_t)Bn * Sn * Dn;              // (B,S,S)

  char* ws = (char*)d_ws;
  bf16_t* srcb = (bf16_t*)ws; ws += (size_t)Bn * Sn * Dn * 2;   // 16 MB
  bf16_t* Wt   = (bf16_t*)ws; ws += (size_t)3 * Dn * Dn * 2;    // 6 MB
  bf16_t* Qp   = (bf16_t*)ws; ws += (size_t)Bn * Sn * Dn * 2;   // 16 MB
  bf16_t* Kp   = (bf16_t*)ws; ws += (size_t)Bn * Sn * Dn * 2;   // 16 MB
  bf16_t* Vt   = (bf16_t*)ws; ws += (size_t)Bn * Sn * Dn * 2;   // 16 MB
  bf16_t* Pb   = (bf16_t*)ws; ws += (size_t)Bn * Sn * Sn * 2;   // 32 MB
  float*  att  = (float*)ws;                                    // 32 MB

  int nsrc = Bn * Sn * Dn;
  cvt_src_kernel<<<(nsrc + 255) / 256, 256, 0, stream>>>(src, srcb, nsrc);
  cvt_wt_kernel<<<dim3(Dn / 32, Dn / 32, 3), 256, 0, stream>>>(Wq, Wk, Wv, Wt);

  qkv_gemm_kernel<<<dim3((Bn * Sn) / TILE_M, Dn / TILE_N, 3), 256, 0, stream>>>(
      srcb, Wt, Qp, Kp, Vt);

  scores_gemm_kernel<<<dim3(Sn / TILE_M, Sn / TILE_N, Bn), 256, 0, stream>>>(Qp, Kp, a);

  softmax_kernel<<<Bn * Sn, 256, 0, stream>>>(a, Pb);

  att_gemm_kernel<<<dim3(Sn / TILE_M, Dn / TILE_N, Bn), 256, 0, stream>>>(Pb, Vt, att);

  addnorm_kernel<<<Bn * Sn, 256, 0, stream>>>(att, src, ln1w, ln1b, ln2w, ln2b, X);
}
